// SINCANet_86620900425895
// MI455X (gfx1250) — compile-verified, matrix operand order checked by reading, not run on hardware
//
#include <hip/hip_runtime.h>
#include <cmath>
#include <cstddef>

// ---------------------------------------------------------------------------
// SINCANet forward for MI455X (gfx1250, wave32, WMMA 16x16x32 f16 -> f32 acc)
// ---------------------------------------------------------------------------

#define DM 256         // D_MODEL
#define NH 8           // heads
#define HD 32          // head dim  (== WMMA K for f16!)
#define LN_EPS 1e-5f
#define PI_F 3.14159265358979323846f

typedef _Float16 half4_t __attribute__((ext_vector_type(4)));
typedef _Float16 half8_t __attribute__((ext_vector_type(8)));
typedef _Float16 v16h_t  __attribute__((ext_vector_type(16)));
typedef float    v8f_t   __attribute__((ext_vector_type(8)));

// Load a 16-lane-major WMMA A/B fragment from a 16-row half region (row
// pointer supplied by caller).  Per-lane: row = lane&15, K-halves =
// (lane>>4)*8 .. +7 and the same +16 (CDNA5 16-bit A/B VGPR layout).
__device__ __forceinline__ v16h_t ldfrag16(const _Float16* p) {
    half8_t lo = *(const half8_t*)p;
    half8_t hi = *(const half8_t*)(p + 16);
    return __builtin_shufflevector(lo, hi, 0,1,2,3,4,5,6,7,8,9,10,11,12,13,14,15);
}

__device__ __forceinline__ v8f_t wmma16(v16h_t a, v16h_t b, v8f_t c) {
    return __builtin_amdgcn_wmma_f32_16x16x32_f16(false, a, false, b, (short)0, c,
                                                  false, false);
}

// ---------------------------------------------------------------------------
// Tiled GEMM: C[M,N] = act(A[M,K] @ B + bias).  A f32 row-major.
// BNK=false: B is [K][N] row-major (linear weights, x @ W)
// BNK=true : B is [N][K] row-major (conv weights viewed [O][I*3*3])
// Tile 128x64x32, 256 threads = 8 waves, each wave 32x32 (2x2 WMMA frags).
// ---------------------------------------------------------------------------
template <bool BNK, int ACT>
__global__ __launch_bounds__(256) void gemm_kernel(
    const float* __restrict__ A, const float* __restrict__ B,
    const float* __restrict__ bias, float* __restrict__ C,
    int M, int N, int K)
{
    __shared__ __align__(16) _Float16 As[128 * 32];
    __shared__ __align__(16) _Float16 Bs[64 * 32];

    const int tid   = threadIdx.x;
    const int lane  = tid & 31;
    const int wave  = tid >> 5;
    const int lrow  = lane & 15;
    const int lhalf = lane >> 4;
    const int wm    = (wave >> 1) * 32;   // 0,32,64,96
    const int wn    = (wave & 1) * 32;    // 0,32
    const int m0    = blockIdx.y * 128;
    const int n0    = blockIdx.x * 64;

    v8f_t acc[2][2] = {};

    for (int k0 = 0; k0 < K; k0 += 32) {
        // ---- stage A (128x32 f32 -> f16), vectorized float4 ----
#pragma unroll
        for (int i = 0; i < 4; ++i) {
            int idx4 = tid + i * 256;           // 0..1023
            int r    = idx4 >> 3;               // 0..127
            int c4   = (idx4 & 7) * 4;          // 0..28
            int gm = m0 + r, gk = k0 + c4;
            half4_t hv;
            if (gm < M && gk + 3 < K) {
                float4 f = *(const float4*)(A + (size_t)gm * K + gk);
                hv = (half4_t){(_Float16)f.x, (_Float16)f.y, (_Float16)f.z, (_Float16)f.w};
                if (i == 0) __builtin_prefetch(A + (size_t)gm * K + gk + 32, 0, 1);
            } else {
#pragma unroll
                for (int u = 0; u < 4; ++u) {
                    float v = (gm < M && (gk + u) < K) ? A[(size_t)gm * K + gk + u] : 0.f;
                    hv[u] = (_Float16)v;
                }
            }
            *(half4_t*)(&As[r * 32 + c4]) = hv;
        }
        // ---- stage B transposed into Bs[n][k] ----
        if (BNK) {
#pragma unroll
            for (int i = 0; i < 2; ++i) {
                int idx4 = tid + i * 256;       // 0..511
                int n    = idx4 >> 3;           // 0..63
                int c4   = (idx4 & 7) * 4;
                int gn = n0 + n, gk = k0 + c4;
                half4_t hv;
                if (gn < N && gk + 3 < K) {
                    float4 f = *(const float4*)(B + (size_t)gn * K + gk);
                    hv = (half4_t){(_Float16)f.x, (_Float16)f.y, (_Float16)f.z, (_Float16)f.w};
                    if (i == 0) __builtin_prefetch(B + (size_t)gn * K + gk + 32, 0, 1);
                } else {
#pragma unroll
                    for (int u = 0; u < 4; ++u) {
                        float v = (gn < N && (gk + u) < K) ? B[(size_t)gn * K + gk + u] : 0.f;
                        hv[u] = (_Float16)v;
                    }
                }
                *(half4_t*)(&Bs[n * 32 + c4]) = hv;
            }
        } else {
#pragma unroll
            for (int i = 0; i < 8; ++i) {
                int idx = tid + i * 256;        // 0..2047
                int n   = idx & 63;
                int k   = idx >> 6;
                int gn = n0 + n, gk = k0 + k;
                float v = (gk < K && gn < N) ? B[(size_t)gk * N + gn] : 0.f;
                if (i == 0 && gk + 32 < K && gn < N)
                    __builtin_prefetch(B + (size_t)(gk + 32) * N + gn, 0, 1);
                Bs[n * 32 + k] = (_Float16)v;
            }
        }
        __syncthreads();

        v16h_t a0 = ldfrag16(&As[(wm +      lrow) * 32 + lhalf * 8]);
        v16h_t a1 = ldfrag16(&As[(wm + 16 + lrow) * 32 + lhalf * 8]);
        v16h_t b0 = ldfrag16(&Bs[(wn +      lrow) * 32 + lhalf * 8]);
        v16h_t b1 = ldfrag16(&Bs[(wn + 16 + lrow) * 32 + lhalf * 8]);

        acc[0][0] = wmma16(a0, b0, acc[0][0]);
        acc[0][1] = wmma16(a0, b1, acc[0][1]);
        acc[1][0] = wmma16(a1, b0, acc[1][0]);
        acc[1][1] = wmma16(a1, b1, acc[1][1]);
        __syncthreads();
    }

    // epilogue: D-fragment layout row = e + 8*(lane/16), col = lane%16
#pragma unroll
    for (int mi = 0; mi < 2; ++mi)
#pragma unroll
        for (int ni = 0; ni < 2; ++ni)
#pragma unroll
            for (int e = 0; e < 8; ++e) {
                int row = m0 + wm + mi * 16 + e + 8 * lhalf;
                int col = n0 + wn + ni * 16 + lrow;
                if (row < M && col < N) {
                    float v = acc[mi][ni][e] + (bias ? bias[col] : 0.f);
                    if (ACT == 1) v = fmaxf(v, 0.f);
                    C[(size_t)row * N + col] = v;
                }
            }
}

// ---------------------------------------------------------------------------
// Flash attention: one wave per (head, 16-query block).  hd=32 == WMMA K.
// Qh: [NH][M][32] f16 (pre-scaled), Kh: [NH][C][32] f16, Vt: [NH][32][C] f16.
// Out: [M, 256] f32.   Requires M%16==0, C%64==0 (true here: 2048/2048).
// 64-key steps: 4 score WMMAs + 4 P.V WMMAs per softmax-reduction round.
// ---------------------------------------------------------------------------
__global__ __launch_bounds__(32) void flash_attn_kernel(
    const _Float16* __restrict__ Qh, const _Float16* __restrict__ Kh,
    const _Float16* __restrict__ Vt, float* __restrict__ Out, int M, int C)
{
    __shared__ __align__(16) _Float16 P[16 * 64];
    const int lane  = threadIdx.x;
    const int lrow  = lane & 15;
    const int lhalf = lane >> 4;
    const int h     = blockIdx.y;
    const int q0    = blockIdx.x * 16;

    const _Float16* qp = Qh + ((size_t)h * M + (q0 + lrow)) * HD + lhalf * 8;
    v16h_t aq = ldfrag16(qp);

    v8f_t o0 = {}, o1 = {};
    float mrow[8], lsum[8];
#pragma unroll
    for (int e = 0; e < 8; ++e) { mrow[e] = -1e30f; lsum[e] = 0.f; }

    const _Float16* kb = Kh + (size_t)h * C * HD;
    const _Float16* vb = Vt + (size_t)h * HD * C;

    for (int j = 0; j < C; j += 64) {
        // scores for keys j .. j+63 (4 fragments of 16 keys)
        v16h_t bk0 = ldfrag16(kb + (size_t)(j +      lrow) * HD + lhalf * 8);
        v16h_t bk1 = ldfrag16(kb + (size_t)(j + 16 + lrow) * HD + lhalf * 8);
        v16h_t bk2 = ldfrag16(kb + (size_t)(j + 32 + lrow) * HD + lhalf * 8);
        v16h_t bk3 = ldfrag16(kb + (size_t)(j + 48 + lrow) * HD + lhalf * 8);
        __builtin_prefetch(kb + (size_t)(j + 64 + lrow) * HD, 0, 1);
        __builtin_prefetch(vb + (size_t)lrow * C + j + 64, 0, 1);
        v8f_t z = {};
        v8f_t s0 = wmma16(aq, bk0, z);
        v8f_t s1 = wmma16(aq, bk1, z);
        v8f_t s2 = wmma16(aq, bk2, z);
        v8f_t s3 = wmma16(aq, bk3, z);

#pragma unroll
        for (int e = 0; e < 8; ++e) {
            float a0 = s0[e], a1 = s1[e], a2 = s2[e], a3 = s3[e];
            float mx = fmaxf(fmaxf(a0, a1), fmaxf(a2, a3));
            mx = fmaxf(mx, __shfl_xor(mx, 1));
            mx = fmaxf(mx, __shfl_xor(mx, 2));
            mx = fmaxf(mx, __shfl_xor(mx, 4));
            mx = fmaxf(mx, __shfl_xor(mx, 8));
            float mn   = fmaxf(mrow[e], mx);
            float corr = __expf(mrow[e] - mn);
            float p0 = __expf(a0 - mn), p1 = __expf(a1 - mn);
            float p2 = __expf(a2 - mn), p3 = __expf(a3 - mn);
            float rs = (p0 + p1) + (p2 + p3);
            rs += __shfl_xor(rs, 1);
            rs += __shfl_xor(rs, 2);
            rs += __shfl_xor(rs, 4);
            rs += __shfl_xor(rs, 8);
            lsum[e] = lsum[e] * corr + rs;
            mrow[e] = mn;
            o0[e] *= corr; o1[e] *= corr;
            int r = e + 8 * lhalf;
            P[r * 64 +      lrow] = (_Float16)p0;
            P[r * 64 + 16 + lrow] = (_Float16)p1;
            P[r * 64 + 32 + lrow] = (_Float16)p2;
            P[r * 64 + 48 + lrow] = (_Float16)p3;
        }
        __syncthreads();
        // P as two A fragments (keys j..j+31 and j+32..j+63)
        v16h_t pa0 = ldfrag16(&P[lrow * 64 +      lhalf * 8]);
        v16h_t pa1 = ldfrag16(&P[lrow * 64 + 32 + lhalf * 8]);
        // V fragments: cols 0-15 / 16-31 of this head's hd, keys chunks
        v16h_t bv00 = ldfrag16(vb + (size_t)(lrow)      * C + j +      lhalf * 8);
        v16h_t bv01 = ldfrag16(vb + (size_t)(lrow)      * C + j + 32 + lhalf * 8);
        v16h_t bv10 = ldfrag16(vb + (size_t)(16 + lrow) * C + j +      lhalf * 8);
        v16h_t bv11 = ldfrag16(vb + (size_t)(16 + lrow) * C + j + 32 + lhalf * 8);
        o0 = wmma16(pa0, bv00, o0);
        o0 = wmma16(pa1, bv01, o0);
        o1 = wmma16(pa0, bv10, o1);
        o1 = wmma16(pa1, bv11, o1);
        __syncthreads();
    }

#pragma unroll
    for (int e = 0; e < 8; ++e) {
        float inv = 1.0f / lsum[e];
        int row = q0 + e + 8 * lhalf;
        Out[(size_t)row * DM + h * HD +      lrow] = o0[e] * inv;
        Out[(size_t)row * DM + h * HD + 16 + lrow] = o1[e] * inv;
    }
}

// ---------------------------------------------------------------------------
// Small elementwise / reduction kernels
// ---------------------------------------------------------------------------
__global__ void zero_kernel(float* p, int n) {
    int i = blockIdx.x * blockDim.x + threadIdx.x;
    if (i < n) p[i] = 0.f;
}

__global__ __launch_bounds__(256) void coord_stats_kernel(
    const float* __restrict__ c, int N, float* __restrict__ stats)
{
    __shared__ float red[256];
    int t = threadIdx.x;
    float am = 0.f, mn0 = 3e38f, mx0 = -3e38f, mn1 = 3e38f, mx1 = -3e38f;
    for (int i = t; i < N; i += 256) {
        float x = c[2 * i], y = c[2 * i + 1];
        am  = fmaxf(am, fmaxf(fabsf(x), fabsf(y)));
        mn0 = fminf(mn0, x); mx0 = fmaxf(mx0, x);
        mn1 = fminf(mn1, y); mx1 = fmaxf(mx1, y);
    }
    float vals[5] = {am, mn0, mx0, mn1, mx1};
    for (int s = 0; s < 5; ++s) {
        red[t] = vals[s]; __syncthreads();
        for (int w = 128; w > 0; w >>= 1) {
            if (t < w) {
                float a = red[t], b = red[t + w];
                red[t] = (s == 1 || s == 3) ? fminf(a, b) : fmaxf(a, b);
            }
            __syncthreads();
        }
        if (t == 0) stats[s] = red[0];
        __syncthreads();
    }
}

__global__ void posenc_kernel(float* __restrict__ x, const float* __restrict__ coords,
                              const float* __restrict__ stats, int N)
{
    int idx = blockIdx.x * blockDim.x + threadIdx.x;
    if (idx >= N * DM) return;
    int n = idx >> 8, j = idx & 255;
    float denom = stats[0] + 1e-8f;
    int odd = j & 1;
    float cn   = coords[2 * n + odd] / denom;
    float freq = (odd ? (float)j : (float)(j + 1)) * PI_F;
    x[idx] += odd ? cosf(cn * freq) : sinf(cn * freq);
}

__global__ void grid_flat_kernel(const float* __restrict__ c,
                                 const float* __restrict__ stats,
                                 int* __restrict__ flat, int N, int H, int W)
{
    int i = blockIdx.x * blockDim.x + threadIdx.x;
    if (i >= N) return;
    float x = c[2 * i], y = c[2 * i + 1];
    float n0 = (x - stats[1]) / (stats[2] - stats[1] + 1e-8f);
    float n1 = (y - stats[3]) / (stats[4] - stats[3] + 1e-8f);
    int i0 = (int)floorf(n0 * (float)(H - 1));
    int i1 = (int)floorf(n1 * (float)(W - 1));
    i0 = min(max(i0, 0), H - 1);
    i1 = min(max(i1, 0), W - 1);
    flat[i] = i0 * W + i1;
}

__global__ void scatter_add_kernel(const float* __restrict__ x, const int* __restrict__ flat,
                                   float* __restrict__ gsum, float* __restrict__ gcnt, int N)
{
    int idx = blockIdx.x * blockDim.x + threadIdx.x;
    if (idx >= N * DM) return;
    int n = idx >> 8, j = idx & 255;
    atomicAdd(&gsum[(size_t)flat[n] * DM + j], x[idx]);
    if (j == 0) atomicAdd(&gcnt[flat[n]], 1.f);
}

__global__ void grid_div_kernel(const float* __restrict__ gsum, const float* __restrict__ gcnt,
                                float* __restrict__ gridf, int HW)
{
    int idx = blockIdx.x * blockDim.x + threadIdx.x;
    if (idx >= HW * DM) return;
    int p = idx >> 8;
    gridf[idx] = gsum[idx] / fmaxf(gcnt[p], 1.f);
}

// im2col for 3x3 SAME conv, NHWC src [H*W][256] -> dst [H*W][2304], k = i*9+kh*3+kw
__global__ void im2col_kernel(const float* __restrict__ src, float* __restrict__ dst,
                              int H, int W)
{
    int idx = blockIdx.x * blockDim.x + threadIdx.x;
    int KK = DM * 9;
    int total = H * W * KK;
    if (idx >= total) return;
    int p = idx / KK, k = idx - p * KK;
    int i = k / 9, r = k - i * 9;
    int kh = r / 3, kw = r - kh * 3;
    int y = p / W, x = p - y * W;
    int yy = y + kh - 1, xx = x + kw - 1;
    float v = 0.f;
    if (yy >= 0 && yy < H && xx >= 0 && xx < W) v = src[(size_t)(yy * W + xx) * DM + i];
    dst[idx] = v;
}

__device__ __forceinline__ float block_ln(float v, int t, float* red) {
    red[t] = v; __syncthreads();
    for (int s = 128; s > 0; s >>= 1) { if (t < s) red[t] += red[t + s]; __syncthreads(); }
    float mean = red[0] / (float)DM; __syncthreads();
    float dv = v - mean;
    red[t] = dv * dv; __syncthreads();
    for (int s = 128; s > 0; s >>= 1) { if (t < s) red[t] += red[t + s]; __syncthreads(); }
    float var = red[0] / (float)DM; __syncthreads();
    return dv * rsqrtf(var + LN_EPS);
}

// y = LN(x + res) * g + b   (res may be null; y may alias x/res)
__global__ __launch_bounds__(256) void ln_kernel(const float* x, const float* res,
                                                 const float* g, const float* b, float* y)
{
    __shared__ float red[256];
    int row = blockIdx.x, t = threadIdx.x;
    float v = x[(size_t)row * DM + t] + (res ? res[(size_t)row * DM + t] : 0.f);
    float nv = block_ln(v, t, red);
    y[(size_t)row * DM + t] = nv * g[t] + b[t];
}

// y = LN(gfeat[flat[row]] + x) * g + b
__global__ __launch_bounds__(256) void ln_gather_kernel(const float* gfeat, const int* flat,
                                                        const float* x, const float* g,
                                                        const float* b, float* y)
{
    __shared__ float red[256];
    int row = blockIdx.x, t = threadIdx.x;
    float v = gfeat[(size_t)flat[row] * DM + t] + x[(size_t)row * DM + t];
    float nv = block_ln(v, t, red);
    y[(size_t)row * DM + t] = nv * g[t] + b[t];
}

// [M,256] f32 -> per-head f16 [NH][M][32] (scaled)
__global__ void pack_heads_kernel(const float* __restrict__ X, _Float16* __restrict__ out,
                                  int M, float scale)
{
    int idx = blockIdx.x * blockDim.x + threadIdx.x;
    if (idx >= M * DM) return;
    int m = idx >> 8, c = idx & 255;
    int h = c >> 5, k = c & 31;
    out[((size_t)h * M + m) * HD + k] = (_Float16)(X[idx] * scale);
}

// [M,256] f32 -> transposed per-head f16 [NH][32][M]
__global__ void pack_heads_T_kernel(const float* __restrict__ X, _Float16* __restrict__ out,
                                    int M)
{
    int idx = blockIdx.x * blockDim.x + threadIdx.x;
    if (idx >= M * DM) return;
    int m = idx >> 8, c = idx & 255;
    int h = c >> 5, k = c & 31;
    out[((size_t)h * HD + k) * M + m] = (_Float16)X[idx];
}

// ---------------------------------------------------------------------------
// Host-side orchestration
// ---------------------------------------------------------------------------
static inline void launch_gemm(bool bnk, int act, const float* A, const float* B,
                               const float* bias, float* C, int M, int N, int K,
                               hipStream_t s)
{
    dim3 g((N + 63) / 64, (M + 127) / 128), b(256);
    if (bnk) {
        if (act) gemm_kernel<true, 1><<<g, b, 0, s>>>(A, B, bias, C, M, N, K);
        else     gemm_kernel<true, 0><<<g, b, 0, s>>>(A, B, bias, C, M, N, K);
    } else {
        if (act) gemm_kernel<false, 1><<<g, b, 0, s>>>(A, B, bias, C, M, N, K);
        else     gemm_kernel<false, 0><<<g, b, 0, s>>>(A, B, bias, C, M, N, K);
    }
}

static void run_mha(const float* qin, const float* kvin, int M, int C,
                    const float* Wq, const float* bq, const float* Wk, const float* bk,
                    const float* Wv, const float* bv, const float* Wo, const float* bo,
                    float* out, float* Qf, float* Kf, float* Vf,
                    _Float16* qh, _Float16* kh, _Float16* vt, float* attnO,
                    hipStream_t s)
{
    launch_gemm(false, 0, qin,  Wq, bq, Qf, M, DM, DM, s);
    launch_gemm(false, 0, kvin, Wk, bk, Kf, C, DM, DM, s);
    launch_gemm(false, 0, kvin, Wv, bv, Vf, C, DM, DM, s);
    float scale = 1.0f / sqrtf((float)HD);
    pack_heads_kernel  <<<(M * DM + 255) / 256, 256, 0, s>>>(Qf, qh, M, scale);
    pack_heads_kernel  <<<(C * DM + 255) / 256, 256, 0, s>>>(Kf, kh, C, 1.0f);
    pack_heads_T_kernel<<<(C * DM + 255) / 256, 256, 0, s>>>(Vf, vt, C);
    flash_attn_kernel  <<<dim3(M / 16, NH), 32, 0, s>>>(qh, kh, vt, attnO, M, C);
    launch_gemm(false, 0, attnO, Wo, bo, out, M, DM, DM, s);
}

extern "C" void kernel_launch(void* const* d_in, const int* in_sizes, int n_in,
                              void* d_out, int out_size, void* d_ws, size_t ws_size,
                              hipStream_t stream)
{
    // Input order: setup_inputs() dict order, params flattened depth-first in
    // insertion order.
    const float* known_genes = (const float*)d_in[0];
    const float* coords      = (const float*)d_in[1];
    const float* sc_data     = (const float*)d_in[2];
    const float* known_W = (const float*)d_in[3];
    const float* known_b = (const float*)d_in[4];
    const float* sc_W    = (const float*)d_in[5];
    const float* sc_b    = (const float*)d_in[6];
    const float* conv1_W = (const float*)d_in[7];
    const float* conv1_b = (const float*)d_in[8];
    const float* conv2_W = (const float*)d_in[9];
    const float* conv2_b = (const float*)d_in[10];
    const float* conv_ln_g = (const float*)d_in[11];
    const float* conv_ln_b = (const float*)d_in[12];
    // ca: Wq,bq,Wk,bk,Wv,bv,Wo,bo,ln_g,ln_b -> 13..22
    // layer l: base 23 + l*16: Wq,bq,Wk,bk,Wv,bv,Wo,bo,ln1_g,ln1_b,W1,b1,W2,b2,ln2_g,ln2_b
    const float* enc_ln_g = (const float*)d_in[55];
    const float* enc_ln_b = (const float*)d_in[56];
    const float* unk_W = (const float*)d_in[57];
    const float* unk_b = (const float*)d_in[58];
    const float* ref_W = (const float*)d_in[59];
    const float* ref_b = (const float*)d_in[60];

    const int N  = in_sizes[1] / 2;            // 2048 spots
    const int Kg = in_sizes[0] / N;            // 2000 known genes
    const int G  = in_sizes[5] / DM;           // 4000 sc genes
    const int C  = in_sizes[2] / G;            // 2048 cells
    const int F  = in_sizes[34];               // 2048 FFN hidden (b1)
    const int U  = in_sizes[58];               // 2000
    const int Ko = in_sizes[60];               // 2000
    const int H  = (int)(std::sqrt((double)N) * 1.5);  // 67
    const int W  = H;
    const int HW = H * W;
    const int MC = (N > C) ? N : C;

    // ---- workspace layout ----
    char* base = (char*)d_ws;
    size_t off = 0;
    auto alloc = [&](size_t bytes) -> void* {
        off = (off + 255) & ~(size_t)255;
        void* p = base + off;
        off += bytes;
        return p;
    };
    float* stats = (float*)alloc(8 * sizeof(float));
    int*   flat  = (int*)  alloc((size_t)N * sizeof(int));
    float* x     = (float*)alloc((size_t)N * DM * sizeof(float));
    float* t0    = (float*)alloc((size_t)N * DM * sizeof(float));
    float* Qf    = (float*)alloc((size_t)MC * DM * sizeof(float));
    float* Kf    = (float*)alloc((size_t)MC * DM * sizeof(float));
    float* Vf    = (float*)alloc((size_t)MC * DM * sizeof(float));
    _Float16* qh = (_Float16*)alloc((size_t)MC * DM * sizeof(_Float16));
    _Float16* kh = (_Float16*)alloc((size_t)MC * DM * sizeof(_Float16));
    _Float16* vt = (_Float16*)alloc((size_t)MC * DM * sizeof(_Float16));
    float* attnO = (float*)alloc((size_t)N * DM * sizeof(float));
    float* scf   = (float*)alloc((size_t)C * DM * sizeof(float));
    float* gsum  = (float*)alloc((size_t)HW * DM * sizeof(float));
    float* gcnt  = (float*)alloc((size_t)HW * sizeof(float));
    float* gridf = (float*)alloc((size_t)HW * DM * sizeof(float));
    float* Mcol  = (float*)alloc((size_t)HW * DM * 9 * sizeof(float));
    float* g1    = (float*)alloc((size_t)HW * DM * sizeof(float));
    float* g2    = (float*)alloc((size_t)HW * DM * sizeof(float));
    float* ffnh  = (float*)alloc((size_t)N * F * sizeof(float));
    (void)ws_size; (void)n_in; (void)out_size;

    // 1) input projection
    launch_gemm(false, 0, known_genes, known_W, known_b, x, N, DM, Kg, stream);

    // 2) coord stats + positional encoding
    coord_stats_kernel<<<1, 256, 0, stream>>>(coords, N, stats);
    posenc_kernel<<<(N * DM + 255) / 256, 256, 0, stream>>>(x, coords, stats, N);

    // 3) spatial conv module
    grid_flat_kernel<<<(N + 255) / 256, 256, 0, stream>>>(coords, stats, flat, N, H, W);
    zero_kernel<<<(HW * DM + 255) / 256, 256, 0, stream>>>(gsum, HW * DM);
    zero_kernel<<<(HW + 255) / 256, 256, 0, stream>>>(gcnt, HW);
    scatter_add_kernel<<<(N * DM + 255) / 256, 256, 0, stream>>>(x, flat, gsum, gcnt, N);
    grid_div_kernel<<<(HW * DM + 255) / 256, 256, 0, stream>>>(gsum, gcnt, gridf, HW);
    im2col_kernel<<<(HW * DM * 9 + 255) / 256, 256, 0, stream>>>(gridf, Mcol, H, W);
    launch_gemm(true, 1, Mcol, conv1_W, conv1_b, g1, HW, DM, DM * 9, stream);
    im2col_kernel<<<(HW * DM * 9 + 255) / 256, 256, 0, stream>>>(g1, Mcol, H, W);
    launch_gemm(true, 0, Mcol, conv2_W, conv2_b, g2, HW, DM, DM * 9, stream);
    ln_gather_kernel<<<N, 256, 0, stream>>>(g2, flat, x, conv_ln_g, conv_ln_b, x);

    // 4) cross-modal attention (spots query cells)
    launch_gemm(false, 0, sc_data, sc_W, sc_b, scf, C, DM, G, stream);
    {
        const float** p = (const float**)&d_in[13];  // ca weights
        run_mha(x, scf, N, C, p[0], p[1], p[2], p[3], p[4], p[5], p[6], p[7],
                t0, Qf, Kf, Vf, qh, kh, vt, attnO, stream);
        ln_kernel<<<N, 256, 0, stream>>>(t0, x, (const float*)d_in[21],
                                         (const float*)d_in[22], x);
    }

    // 5) transformer encoder layers
    for (int l = 0; l < 2; ++l) {
        int bi = 23 + l * 16;
        const float** p = (const float**)&d_in[bi];
        run_mha(x, x, N, N, p[0], p[1], p[2], p[3], p[4], p[5], p[6], p[7],
                t0, Qf, Kf, Vf, qh, kh, vt, attnO, stream);
        ln_kernel<<<N, 256, 0, stream>>>(t0, x, p[8], p[9], x);
        launch_gemm(false, 1, x, p[10], p[11], ffnh, N, F, DM, stream);
        launch_gemm(false, 0, ffnh, p[12], p[13], t0, N, DM, F, stream);
        ln_kernel<<<N, 256, 0, stream>>>(t0, x, p[14], p[15], x);
    }
    ln_kernel<<<N, 256, 0, stream>>>(x, nullptr, enc_ln_g, enc_ln_b, x);

    // 6) output heads (concatenated: unknown then refined)
    float* out = (float*)d_out;
    launch_gemm(false, 0, x, unk_W, unk_b, out, N, U, DM, stream);
    launch_gemm(false, 0, x, ref_W, ref_b, out + (size_t)N * U, N, Ko, DM, stream);
}
